// LSTMNet_48215302865741
// MI455X (gfx1250) — matrix operand order repaired and checked by reading, not compiled, not run on hardware
//
#include <hip/hip_runtime.h>
#include <hip/hip_bf16.h>
#include <math.h>

// ---------------------------------------------------------------------------
// CDNA5 / gfx1250 implementation of LSTMNet (LSTM scan -> prenet conv -> ResNet18 -> MLP)
// Conv/FC layers: implicit GEMM on v_wmma_f32_16x16x32_bf16 (wave32), with the
// weight (B) tiles staged into LDS by the Tensor Data Mover (tensor_load_to_lds,
// double buffered, s_wait_tensorcnt) and activations (A) gathered via a
// precomputed LDS im2col-offset table.
// ---------------------------------------------------------------------------

#define CONV_TDM 1   // stage B-tiles with the Tensor Data Mover

typedef __attribute__((ext_vector_type(16))) __bf16 v16bf;
typedef __attribute__((ext_vector_type(8)))  float  v8f;
typedef unsigned int v4u __attribute__((ext_vector_type(4)));
typedef int v8i __attribute__((ext_vector_type(8)));
typedef int v4i __attribute__((ext_vector_type(4)));

__device__ __forceinline__ __bf16 f2bf(float f) {
  unsigned u = __builtin_bit_cast(unsigned, f);
  unsigned r = u + 0x7FFFu + ((u >> 16) & 1u);        // round-to-nearest-even
  unsigned short s = (unsigned short)(r >> 16);
  return __builtin_bit_cast(__bf16, s);
}

__device__ __forceinline__ float sigf(float x) { return 1.f / (1.f + __expf(-x)); }

// mode bits
#define MODE_BIAS 1
#define MODE_BN   2
#define MODE_RES  4
#define ACT_NONE  (0 << 4)
#define ACT_RELU  (1 << 4)
#define ACT_ELU   (2 << 4)
#define ACT_LEAKY (3 << 4)
#define ACT_SIG   (4 << 4)

__device__ __forceinline__ float apply_act(float v, int act) {
  switch (act) {
    case 1: return fmaxf(v, 0.f);
    case 2: return v > 0.f ? v : (__expf(v) - 1.f);       // ELU(alpha=1)
    case 3: return v > 0.f ? v : 0.01f * v;               // leaky_relu(0.01)
    case 4: return sigf(v);
    default: return v;
  }
}

#if CONV_TDM
// ---------------------------------------------------------------------------
// TDM: DMA a 2D bf16 tile [nRows x kElems] (row stride = rowStrideElems) from
// global into LDS at byte offset ldsOff (rows stored contiguously).
// D# bitfields per CDNA5 ISA.
// ---------------------------------------------------------------------------
__device__ __forceinline__ void tdm_load_tile_bf16(const __bf16* gsrc, unsigned ldsOff,
                                                   int kElems, int nRows,
                                                   int rowStrideElems) {
  unsigned long long ga = (unsigned long long)(uintptr_t)gsrc;
  v4u g0;
  g0.x = 1u;                                            // count=1, user, no gather
  g0.y = ldsOff;                                        // lds_addr (bytes)
  g0.z = (unsigned)ga;                                  // global_addr[31:0]
  g0.w = (unsigned)((ga >> 32) & 0x1FFFFFFu) | (2u << 30);  // addr[56:32] | type=2
  unsigned td0 = (unsigned)rowStrideElems;              // tensor_dim0 (elements)
  unsigned td1 = (unsigned)nRows;                       // tensor_dim1
  unsigned long long s0 = (unsigned long long)rowStrideElems;  // dim0 stride
  v8i g1;
  g1[0] = (int)(1u << 16);                              // wg_mask=0 | data_size=1 (2B)
  g1[1] = (int)((td0 & 0xFFFFu) << 16);                 // atomic_addr=0 | td0[15:0]
  g1[2] = (int)((td0 >> 16) | ((td1 & 0xFFFFu) << 16)); // td0[31:16] | td1[15:0]
  g1[3] = (int)((td1 >> 16) | ((unsigned)kElems << 16));// td1[31:16] | tile_dim0
  g1[4] = (int)(unsigned)nRows;                         // tile_dim1 | tile_dim2=0
  g1[5] = (int)(unsigned)s0;                            // stride0[31:0]
  g1[6] = (int)((unsigned)(s0 >> 32) & 0xFFFFu);        // stride0[47:32] | stride1[15:0]=0
  g1[7] = 0;                                            // stride1[47:16]=0
  v4i gz = {0, 0, 0, 0};
#if __clang_major__ >= 23
  v8i gz8 = {0, 0, 0, 0, 0, 0, 0, 0};
  __builtin_amdgcn_tensor_load_to_lds(g0, g1, gz, gz, gz8, 0);
#else
  __builtin_amdgcn_tensor_load_to_lds(g0, g1, gz, gz, 0);
#endif
}
#endif  // CONV_TDM

// ---------------------------------------------------------------------------
// Weight pack: fp32 [N,K] (OIHW is n-major, k=(ci*KH+kh)*KW+kw) ->
// bf16 [Npad32, Kpad64], zero filled.  Lane B-fragments are 32B vectors.
// ---------------------------------------------------------------------------
__global__ void pack_w_bf16_k(const float* __restrict__ w, __bf16* __restrict__ wp,
                              int N, int K, int Npad, int Kpad) {
  int idx = blockIdx.x * 256 + threadIdx.x;
  int total = Npad * Kpad;
  if (idx >= total) return;
  int n = idx / Kpad, k = idx - n * Kpad;
  float v = (n < N && k < K) ? w[(size_t)n * K + k] : 0.f;
  wp[idx] = f2bf(v);
}

// ---------------------------------------------------------------------------
// xg[t,b,g] = sum_i x[b,i,t] * W_ih[g,i] + b_ih[g] + b_hh[g]
// ---------------------------------------------------------------------------
__global__ void xg_proj_k(const float* __restrict__ x, const float* __restrict__ Wih,
                          const float* __restrict__ bih, const float* __restrict__ bhh,
                          float* __restrict__ xg, int total) {
  int idx = blockIdx.x * 256 + threadIdx.x;
  if (idx >= total) return;
  const int T = 224 * 224;
  int g = idx % 48;
  int r = idx / 48;
  int b = r % 32;
  int t = r / 32;
  float x0 = x[((size_t)b * 3 + 0) * T + t];
  float x1 = x[((size_t)b * 3 + 1) * T + t];
  float x2 = x[((size_t)b * 3 + 2) * T + t];
  xg[idx] = x0 * Wih[g * 3 + 0] + x1 * Wih[g * 3 + 1] + x2 * Wih[g * 3 + 2]
          + bih[g] + bhh[g];
}

// ---------------------------------------------------------------------------
// Persistent LSTM scan: one workgroup, 384 threads = (b,j) in [32]x[12].
// ---------------------------------------------------------------------------
__global__ __launch_bounds__(384) void lstm_scan_k(const float* __restrict__ xg,
                                                   const float* __restrict__ Whh,
                                                   float* __restrict__ y) {
  __shared__ float sh[32][12];
  const int tid = threadIdx.x;
  const int b = tid / 12;
  const int j = tid - b * 12;
  float wi[12], wf[12], wg[12], wo[12];
#pragma unroll
  for (int k = 0; k < 12; ++k) {
    wi[k] = Whh[(j)      * 12 + k];
    wf[k] = Whh[(12 + j) * 12 + k];
    wg[k] = Whh[(24 + j) * 12 + k];
    wo[k] = Whh[(36 + j) * 12 + k];
  }
  float c = 0.f;
  sh[b][j] = 0.f;
  __syncthreads();
  const int T = 224 * 224;
  for (int t = 0; t < T; ++t) {
    if ((t & 63) == 0)  // gfx1250 global_prefetch_b8 of upcoming gate rows
      __builtin_prefetch(xg + (size_t)(t + 64) * 1536 + tid, 0, 1);
    float hl[12];
#pragma unroll
    for (int k = 0; k < 12; ++k) hl[k] = sh[b][k];
    const float* g4 = xg + (size_t)t * 1536 + b * 48;
    float gi = g4[j], gf = g4[12 + j], gg = g4[24 + j], go = g4[36 + j];
#pragma unroll
    for (int k = 0; k < 12; ++k) {
      gi += hl[k] * wi[k];
      gf += hl[k] * wf[k];
      gg += hl[k] * wg[k];
      go += hl[k] * wo[k];
    }
    c = sigf(gf) * c + sigf(gi) * tanhf(gg);
    float h = sigf(go) * tanhf(c);
    __syncthreads();
    sh[b][j] = h;
    y[((size_t)b * 12 + j) * T + t] = h;
    __syncthreads();
  }
}

// ---------------------------------------------------------------------------
// Implicit-GEMM conv (also all FC layers with H=W=KH=KW=1).
//   M = B*OH*OW, N = Cout, K = Cin*KH*KW (Kpad multiple of 64).
//   128 threads = 4 waves; each wave owns 16(M) x 32(N); per 64-wide K step
//   it issues 2 sub-steps x 2 WMMAs reusing one A fragment each.
//   B tile (32n x 64k bf16) is TDM-DMA'd to LDS, double buffered: wave0 issues
//   the next tile, s_wait_tensorcnt(1) retires the current, block barriers.
//   A gather uses an LDS table of precombined im2col offsets.
// A-frag: lane&15 = M row; lane half selects K 0..15 vs
//   16..31 per sub-step, same as B. C/D: lane&15 = N col, lane half selects M rows r vs r+8.
// ---------------------------------------------------------------------------
__global__ __launch_bounds__(128) void conv_wmma_k(
    const float* __restrict__ in, const __bf16* __restrict__ wp,
    const float* __restrict__ bias, const float* __restrict__ bng,
    const float* __restrict__ bnb, const float* __restrict__ bnm,
    const float* __restrict__ bnv, const float* __restrict__ resid,
    float* __restrict__ out,
    int Bn, int Cin, int H, int W, int Cout, int KH, int KW,
    int stride, int pad, int OH, int OW, int Kreal, int Kpad, int mode) {
  __shared__ __bf16 sB[2][32 * 64];   // double-buffered B tile (rows of 64 k)
  __shared__ int s_off[64];           // ci*H*W + kh*W + kw  (per k in step)
  __shared__ int s_khw[64];           // (kh<<16)|kw, 0x7FFF7FFF = invalid k
  const int lane = threadIdx.x & 31;
  const int wv = threadIdx.x >> 5;
  const int half = lane >> 4;
  const int M = Bn * OH * OW;
  const int mBase = blockIdx.x * 64 + wv * 16;
  const int nBase = blockIdx.y * 32;
  const int HWin = H * W;
  const int KHW = KH * KW;

  // A-fragment row for this lane
  const int mrow = mBase + (lane & 15);
  const bool mvalid = (mrow < M);
  const int mm = mvalid ? mrow : 0;
  const int bi = mm / (OH * OW);
  const int sp = mm - bi * (OH * OW);
  const int oh = sp / OW, ow = sp - (sp / OW) * OW;
  const int ih0 = oh * stride - pad;
  const int iw0 = ow * stride - pad;
  const float* inBase = in + (size_t)bi * Cin * HWin + (ih0 * W + iw0);

  const int n0 = nBase + (lane & 15);
  const int n1 = n0 + 16;
  const int nloc = lane & 15;

  v8f acc0 = {};
  v8f acc1 = {};
  const int nsteps = Kpad >> 6;
  const __bf16* wtile = wp + (size_t)nBase * Kpad;

#if CONV_TDM
  if (wv == 0)  // prologue: DMA B tile for k-step 0 into buffer 0
    tdm_load_tile_bf16(wtile, (unsigned)(uintptr_t)&sB[0][0], 64, 32, Kpad);
#endif

  for (int s = 0; s < nsteps; ++s) {
    const int k0 = s << 6;
    __syncthreads();  // previous step's LDS consumers are done
    if (threadIdx.x < 64) {
      int k = k0 + threadIdx.x;
      int off = 0, khw = 0x7FFF7FFF;
      if (k < Kreal) {
        int ci = k / KHW;
        int r = k - ci * KHW;
        int kh = r / KW;
        int kw = r - kh * KW;
        off = ci * HWin + kh * W + kw;
        khw = (kh << 16) | kw;
      }
      s_off[threadIdx.x] = off;
      s_khw[threadIdx.x] = khw;
    }
#if CONV_TDM
    if (wv == 0) {
      if (s + 1 < nsteps) {
        tdm_load_tile_bf16(wtile + (size_t)(k0 + 64),
                           (unsigned)(uintptr_t)&sB[(s + 1) & 1][0], 64, 32, Kpad);
        __builtin_amdgcn_s_wait_tensorcnt(1);  // retire current tile's DMA
      } else {
        __builtin_amdgcn_s_wait_tensorcnt(0);
      }
    }
#endif
    __syncthreads();  // tables + current B tile visible to all waves

#if CONV_TDM
    const __bf16* sBr = &sB[s & 1][0];
#endif
#pragma unroll
    for (int sub = 0; sub < 2; ++sub) {
      v16bf a;
#pragma unroll
      for (int j = 0; j < 16; ++j) {
        int koff = sub * 32 + half * 16 + j;  // lane half selects K 0..15 / 16..31
        int khw = s_khw[koff];
        int kh = khw >> 16, kw = khw & 0xFFFF;
        int ih = ih0 + kh, iw = iw0 + kw;
        float v = 0.f;
        if (mvalid && (unsigned)ih < (unsigned)H && (unsigned)iw < (unsigned)W)
          v = inBase[s_off[koff]];
        a[j] = f2bf(v);
      }
      const int kb = sub * 32 + half * 16;  // lane half selects K 0..15 / 16..31
#if CONV_TDM
      v16bf b0v = *(const v16bf*)(sBr + nloc * 64 + kb);
      v16bf b1v = *(const v16bf*)(sBr + (nloc + 16) * 64 + kb);
#else
      v16bf b0v = *(const v16bf*)(wp + (size_t)n0 * Kpad + k0 + kb);
      v16bf b1v = *(const v16bf*)(wp + (size_t)n1 * Kpad + k0 + kb);
#endif
      acc0 = __builtin_amdgcn_wmma_f32_16x16x32_bf16(false, a, false, b0v,
                                                     (short)0, acc0, false, false);
      acc1 = __builtin_amdgcn_wmma_f32_16x16x32_bf16(false, a, false, b1v,
                                                     (short)0, acc1, false, false);
    }
  }

  // --- fused epilogue: bias / BN / residual / activation ---
  const bool hasBias = mode & MODE_BIAS;
  const bool hasBN = mode & MODE_BN;
  const bool hasRes = mode & MODE_RES;
  const int act = mode >> 4;

  float add0 = 0.f, scl0 = 1.f, sh0 = 0.f;
  float add1 = 0.f, scl1 = 1.f, sh1 = 0.f;
  if (n0 < Cout) {
    if (hasBias) add0 = bias[n0];
    if (hasBN) {
      float s = bng[n0] * rsqrtf(bnv[n0] + 1e-5f);
      scl0 = s; sh0 = bnb[n0] - bnm[n0] * s;
    }
  }
  if (n1 < Cout) {
    if (hasBias) add1 = bias[n1];
    if (hasBN) {
      float s = bng[n1] * rsqrtf(bnv[n1] + 1e-5f);
      scl1 = s; sh1 = bnb[n1] - bnm[n1] * s;
    }
  }

#pragma unroll
  for (int r = 0; r < 8; ++r) {
    int m = mBase + half * 8 + r;  // D-frag row for this lane/vgpr
    if (m >= M) continue;
    int b2 = m / (OH * OW);
    int sp2 = m - b2 * (OH * OW);
    int oh2 = sp2 / OW, ow2 = sp2 - (sp2 / OW) * OW;
    if (n0 < Cout) {
      size_t idx = (((size_t)b2 * Cout + n0) * OH + oh2) * OW + ow2;
      float v = (acc0[r] + add0) * scl0 + sh0;
      if (hasRes) v += resid[idx];
      out[idx] = apply_act(v, act);
    }
    if (n1 < Cout) {
      size_t idx = (((size_t)b2 * Cout + n1) * OH + oh2) * OW + ow2;
      float v = (acc1[r] + add1) * scl1 + sh1;
      if (hasRes) v += resid[idx];
      out[idx] = apply_act(v, act);
    }
  }
}

// ---------------------------------------------------------------------------
// maxpool 3x3 stride2 pad1  and  global average pool
// ---------------------------------------------------------------------------
__global__ void maxpool3x3s2_k(const float* __restrict__ in, float* __restrict__ out,
                               int C, int H, int W, int OH, int OW, int total) {
  int idx = blockIdx.x * 256 + threadIdx.x;
  if (idx >= total) return;
  int ow = idx % OW;
  int t = idx / OW;
  int oh = t % OH; t /= OH;
  int c = t % C;
  int b = t / C;
  float m = -INFINITY;
  int ih0 = oh * 2 - 1, iw0 = ow * 2 - 1;
#pragma unroll
  for (int r = 0; r < 3; ++r) {
    int ih = ih0 + r;
    if ((unsigned)ih >= (unsigned)H) continue;
#pragma unroll
    for (int s = 0; s < 3; ++s) {
      int iw = iw0 + s;
      if ((unsigned)iw >= (unsigned)W) continue;
      m = fmaxf(m, in[(((size_t)b * C + c) * H + ih) * W + iw]);
    }
  }
  out[idx] = m;
}

__global__ void avgpool_k(const float* __restrict__ in, float* __restrict__ out,
                          int HW, int total) {
  int idx = blockIdx.x * 256 + threadIdx.x;
  if (idx >= total) return;
  const float* p = in + (size_t)idx * HW;
  float s = 0.f;
  for (int i = 0; i < HW; ++i) s += p[i];
  out[idx] = s / (float)HW;
}

// ---------------------------------------------------------------------------
// Host orchestration
// ---------------------------------------------------------------------------
static const __bf16* pack_w(hipStream_t st, const float* w, int N, int K, char*& cur) {
  int Npad = (N + 31) & ~31;
  int Kpad = (K + 63) & ~63;
  __bf16* wp = (__bf16*)cur;
  size_t bytes = (size_t)Npad * Kpad * sizeof(__bf16);
  cur += (bytes + 255) & ~(size_t)255;
  int total = Npad * Kpad;
  pack_w_bf16_k<<<(total + 255) / 256, 256, 0, st>>>(w, wp, N, K, Npad, Kpad);
  return wp;
}

static void launch_conv(hipStream_t st, const float* in, const __bf16* wp,
                        const float* bias, const float* g, const float* b,
                        const float* m, const float* v, const float* resid,
                        float* out, int Bn, int Cin, int H, int W, int Cout,
                        int KH, int KW, int stride, int pad, int mode) {
  int OH = (H + 2 * pad - KH) / stride + 1;
  int OW = (W + 2 * pad - KW) / stride + 1;
  int Kreal = Cin * KH * KW;
  int Kpad = (Kreal + 63) & ~63;
  int M = Bn * OH * OW;
  int Npad = (Cout + 31) & ~31;
  dim3 gr((M + 63) / 64, Npad / 32);
  conv_wmma_k<<<gr, 128, 0, st>>>(in, wp, bias, g, b, m, v, resid, out, Bn, Cin,
                                  H, W, Cout, KH, KW, stride, pad, OH, OW, Kreal,
                                  Kpad, mode);
}

struct BlkIdx { int c1, bn1, c2, bn2, dc, dbn; };

extern "C" void kernel_launch(void* const* d_in, const int* in_sizes, int n_in,
                              void* d_out, int out_size, void* d_ws, size_t ws_size,
                              hipStream_t stream) {
  (void)in_sizes; (void)n_in; (void)out_size; (void)ws_size;
  const float* X   = (const float*)d_in[0];
  const float* Wih = (const float*)d_in[1];
  const float* Whh = (const float*)d_in[2];
  const float* bih = (const float*)d_in[3];
  const float* bhh = (const float*)d_in[4];
  auto F = [&](int i) { return (const float*)d_in[i]; };

  // flat input index map (setup_inputs dict order)
  static const BlkIdx BL[8] = {
      {12, 13, 17, 18, -1, -1},  {22, 23, 27, 28, -1, -1},
      {32, 33, 37, 38, 42, 43},  {47, 48, 52, 53, -1, -1},
      {57, 58, 62, 63, 67, 68},  {72, 73, 77, 78, -1, -1},
      {82, 83, 87, 88, 92, 93},  {97, 98, 102, 103, -1, -1}};
  static const int CINB[8]  = {64, 64, 64, 128, 128, 128, 256, 256};
  static const int COUTB[8] = {64, 64, 128, 128, 256, 256, 512, 512};
  static const int STRB[8]  = {1, 1, 2, 1, 2, 1, 2, 1};
  static const int HINB[8]  = {56, 56, 56, 28, 28, 28, 14, 14};

  // workspace layout (≈ 641 MB): xg | A0 | A1 | A2 | packed bf16 weights
  const int T = 224 * 224;
  char* ws = (char*)d_ws;
  float* XG = (float*)ws;                               // 50176*32*48*4
  size_t xgBytes = (size_t)T * 32 * 48 * 4;             // 308,281,344
  size_t ABUF = (size_t)32 * 64 * 112 * 112 * 4;        // 102,760,448
  float* A0 = (float*)(ws + xgBytes);
  float* A1 = (float*)(ws + xgBytes + ABUF);
  float* A2 = (float*)(ws + xgBytes + 2 * ABUF);
  char* wcur = ws + xgBytes + 3 * ABUF;

  // ---- pack all weights to bf16 [Npad32, Kpad64] ----
  const __bf16* wp_pre  = pack_w(stream, F(5), 3, 12 * 9, wcur);
  const __bf16* wp_stem = pack_w(stream, F(7), 64, 3 * 49, wcur);
  const __bf16* wp_c1[8]; const __bf16* wp_c2[8]; const __bf16* wp_dc[8];
  for (int i = 0; i < 8; ++i) {
    wp_c1[i] = pack_w(stream, F(BL[i].c1), COUTB[i], CINB[i] * 9, wcur);
    wp_c2[i] = pack_w(stream, F(BL[i].c2), COUTB[i], COUTB[i] * 9, wcur);
    wp_dc[i] = (BL[i].dc >= 0) ? pack_w(stream, F(BL[i].dc), COUTB[i], CINB[i], wcur)
                               : nullptr;
  }
  const __bf16* wp_fc = pack_w(stream, F(107), 1000, 512, wcur);
  const __bf16* wp_m1 = pack_w(stream, F(109), 512, 1000, wcur);
  const __bf16* wp_m2 = pack_w(stream, F(115), 128, 512, wcur);
  const __bf16* wp_m3 = pack_w(stream, F(121), 128, 128, wcur);
  const __bf16* wp_m4 = pack_w(stream, F(123), 2, 128, wcur);

  // ---- LSTM: streamed input projection, then persistent serial scan ----
  {
    int total = T * 32 * 48;
    xg_proj_k<<<(total + 255) / 256, 256, 0, stream>>>(X, Wih, bih, bhh, XG, total);
    lstm_scan_k<<<1, 384, 0, stream>>>(XG, Whh, A0);    // A0 = [32,12,224,224]
  }

  // ---- prenet conv 12->3 (bias + ELU) ----
  launch_conv(stream, A0, wp_pre, F(6), nullptr, nullptr, nullptr, nullptr, nullptr,
              A1, 32, 12, 224, 224, 3, 3, 3, 1, 1, MODE_BIAS | ACT_ELU);

  // ---- ResNet stem 3->64 7x7 s2 (BN + ReLU), maxpool 3x3 s2 ----
  launch_conv(stream, A1, wp_stem, nullptr, F(8), F(9), F(10), F(11), nullptr,
              A0, 32, 3, 224, 224, 64, 7, 7, 2, 3, MODE_BN | ACT_RELU);
  {
    int total = 32 * 64 * 56 * 56;
    maxpool3x3s2_k<<<(total + 255) / 256, 256, 0, stream>>>(A0, A1, 64, 112, 112,
                                                            56, 56, total);
  }

  // ---- 8 residual blocks (conv+BN fused; residual+ReLU fused into c2) ----
  float* bufs[3] = {A0, A1, A2};
  float* cur = A1;
  auto bid = [&](float* p) { return p == A0 ? 0 : (p == A1 ? 1 : 2); };
  for (int i = 0; i < 8; ++i) {
    int Hin = HINB[i], st = STRB[i];
    float* o1 = bufs[(bid(cur) + 1) % 3];
    float* o2 = bufs[(bid(cur) + 2) % 3];
    if (BL[i].dc >= 0) {
      // ident = BN(dc(cur)) -> o2 ; z1 = relu(BN(c1(cur))) -> o1 ;
      // cur = relu(BN(c2(z1)) + ident)
      launch_conv(stream, cur, wp_dc[i], nullptr, F(BL[i].dbn), F(BL[i].dbn + 1),
                  F(BL[i].dbn + 2), F(BL[i].dbn + 3), nullptr, o2, 32, CINB[i],
                  Hin, Hin, COUTB[i], 1, 1, st, 0, MODE_BN | ACT_NONE);
      launch_conv(stream, cur, wp_c1[i], nullptr, F(BL[i].bn1), F(BL[i].bn1 + 1),
                  F(BL[i].bn1 + 2), F(BL[i].bn1 + 3), nullptr, o1, 32, CINB[i],
                  Hin, Hin, COUTB[i], 3, 3, st, 1, MODE_BN | ACT_RELU);
      int Ho = Hin / st;
      launch_conv(stream, o1, wp_c2[i], nullptr, F(BL[i].bn2), F(BL[i].bn2 + 1),
                  F(BL[i].bn2 + 2), F(BL[i].bn2 + 3), o2, cur, 32, COUTB[i],
                  Ho, Ho, COUTB[i], 3, 3, 1, 1, MODE_BN | MODE_RES | ACT_RELU);
      // cur unchanged
    } else {
      launch_conv(stream, cur, wp_c1[i], nullptr, F(BL[i].bn1), F(BL[i].bn1 + 1),
                  F(BL[i].bn1 + 2), F(BL[i].bn1 + 3), nullptr, o1, 32, CINB[i],
                  Hin, Hin, COUTB[i], 3, 3, 1, 1, MODE_BN | ACT_RELU);
      launch_conv(stream, o1, wp_c2[i], nullptr, F(BL[i].bn2), F(BL[i].bn2 + 1),
                  F(BL[i].bn2 + 2), F(BL[i].bn2 + 3), cur, o2, 32, COUTB[i],
                  Hin, Hin, COUTB[i], 3, 3, 1, 1, MODE_BN | MODE_RES | ACT_RELU);
      cur = o2;
    }
  }

  // ---- global avgpool [32,512,7,7] -> [32,512] ----
  float* P = bufs[(bid(cur) + 1) % 3];
  float* Q = bufs[(bid(cur) + 2) % 3];
  {
    int total = 32 * 512;
    avgpool_k<<<(total + 255) / 256, 256, 0, stream>>>(cur, P, 49, total);
  }

  // ---- FC + MLP head (same WMMA GEMM kernel, 1x1 spatial) ----
  launch_conv(stream, P, wp_fc, F(108), nullptr, nullptr, nullptr, nullptr, nullptr,
              Q, 32, 512, 1, 1, 1000, 1, 1, 1, 0, MODE_BIAS | ACT_NONE);
  launch_conv(stream, Q, wp_m1, F(110), F(111), F(112), F(113), F(114), nullptr,
              P, 32, 1000, 1, 1, 512, 1, 1, 1, 0, MODE_BIAS | MODE_BN | ACT_LEAKY);
  launch_conv(stream, P, wp_m2, F(116), F(117), F(118), F(119), F(120), nullptr,
              Q, 32, 512, 1, 1, 128, 1, 1, 1, 0, MODE_BIAS | MODE_BN | ACT_LEAKY);
  launch_conv(stream, Q, wp_m3, F(122), nullptr, nullptr, nullptr, nullptr, nullptr,
              P, 32, 128, 1, 1, 128, 1, 1, 1, 0, MODE_BIAS | ACT_LEAKY);
  launch_conv(stream, P, wp_m4, F(124), nullptr, nullptr, nullptr, nullptr, nullptr,
              (float*)d_out, 32, 128, 1, 1, 2, 1, 1, 1, 0, MODE_BIAS | ACT_SIG);
}
